// SE3ProjectedConjugatedCSPNet_85109071937591
// MI455X (gfx1250) — matrix operand order repaired and checked by reading, not compiled, not run on hardware
//
#include <hip/hip_runtime.h>

// ---------------------------------------------------------------------------
// Problem constants (from reference)
// ---------------------------------------------------------------------------
#define NN 10000          // nodes
#define NE 100000         // edges
#define NG 64             // graphs
#define HID 512
#define KPAD1 1952        // 1926 padded up to multiple of 32 (61 chunks)
#define TAILW 416         // 6 (lattice) + 384 (emb) padded to multiple of 32

#define BM 128
#define BN 128
#define BK 32
#define LDST 40           // LDS row stride in halves (80B, 16B-aligned, conflict-free)

typedef __attribute__((ext_vector_type(16))) _Float16 v16h;
typedef __attribute__((ext_vector_type(8)))  float    v8f;

union HFrag { v16h v; uint4 q[2]; };
union FAcc  { v8f  v; float f[8]; };
union H8    { _Float16 h[8]; uint4 q; };

__device__ __forceinline__ float silu_f(float x) {
    // x * sigmoid(x); fast v_rcp_f32 instead of the precise-division sequence
    return x * __builtin_amdgcn_rcpf(1.0f + __expf(-x));
}

// ---------------------------------------------------------------------------
// Small prep kernels
// ---------------------------------------------------------------------------
__global__ void zerof_kernel(float* p, long long n) {
    long long i = (long long)blockIdx.x * blockDim.x + threadIdx.x;
    if (i < n) p[i] = 0.0f;
}

// W: [K][512] f32  ->  WT: [512][Kpad] f16 (transposed, zero-padded K)
__global__ void packw_kernel(const float* __restrict__ W, _Float16* __restrict__ WT,
                             int K, int Kpad) {
    long long i = (long long)blockIdx.x * blockDim.x + threadIdx.x;
    long long total = 512LL * Kpad;
    if (i >= total) return;
    int n = (int)(i / Kpad);
    int k = (int)(i % Kpad);
    WT[i] = (k < K) ? (_Float16)W[(size_t)k * 512 + n] : (_Float16)0.0f;
}

// lattice_params_to_matrix: lattices [64][6] -> L [64][3][3] row-major
__global__ void lattice_kernel(const float* __restrict__ lat, float* __restrict__ L) {
    int g = blockIdx.x * blockDim.x + threadIdx.x;
    if (g >= NG) return;
    const float* p = lat + g * 6;
    float a = p[0], b = p[1], c = p[2];
    const float d2r = 3.14159265358979323846f / 180.0f;
    float c0 = cosf(p[3] * d2r), c1 = cosf(p[4] * d2r), c2 = cosf(p[5] * d2r);
    float s0 = sinf(p[3] * d2r), s1 = sinf(p[4] * d2r);
    float val = (c0 * c1 - c2) / (s0 * s1);
    val = fminf(1.0f, fmaxf(-1.0f, val));
    float gs = acosf(val);
    float* o = L + g * 9;
    o[0] = a * s1;             o[1] = 0.0f;              o[2] = a * c1;
    o[3] = -b * s0 * cosf(gs); o[4] = b * s0 * sinf(gs); o[5] = b * c0;
    o[6] = 0.0f;               o[7] = 0.0f;              o[8] = c;
}

// LayerNorm: one wave32 per node row (512 cols). Writes f16 into nodecat[:, 0:512].
__global__ __launch_bounds__(256)
void layernorm_kernel(const float* __restrict__ x, const float* __restrict__ gamma,
                      const float* __restrict__ beta, _Float16* __restrict__ nodecat,
                      int n) {
    int wid  = threadIdx.x >> 5;
    int lane = threadIdx.x & 31;
    int row  = blockIdx.x * 8 + wid;
    if (row >= n) return;
    const float* xr = x + (size_t)row * HID;
    float v[16], s = 0.0f, ss = 0.0f;
    #pragma unroll
    for (int i = 0; i < 16; ++i) {
        float t = xr[lane + i * 32];
        v[i] = t; s += t; ss += t * t;
    }
    #pragma unroll
    for (int o = 16; o > 0; o >>= 1) {
        s  += __shfl_xor(s,  o, 32);
        ss += __shfl_xor(ss, o, 32);
    }
    float mu  = s * (1.0f / HID);
    float var = ss * (1.0f / HID) - mu * mu;
    float inv = rsqrtf(var + 1e-5f);
    _Float16* out = nodecat + (size_t)row * 1024;
    #pragma unroll
    for (int i = 0; i < 16; ++i) {
        int c = lane + i * 32;
        out[c] = (_Float16)((v[i] - mu) * inv * gamma[c] + beta[c]);
    }
}

// Tail features per edge: [lat_e(6) | sin(192) | cos(192) | pad(26)] -> f16 [NE][416]
__global__ void tail_kernel(const float* __restrict__ frac, const float* __restrict__ lat,
                            const float* __restrict__ L, const int* __restrict__ e2g,
                            _Float16* __restrict__ tail) {
    long long i = (long long)blockIdx.x * blockDim.x + threadIdx.x;
    if (i >= (long long)NE * TAILW) return;
    int e = (int)(i / TAILW);
    int t = (int)(i % TAILW);
    int g = e2g[e];
    _Float16 out;
    if (t < 6) {
        out = (_Float16)lat[g * 6 + t];
    } else if (t < 390) {
        int s   = (t < 198) ? (t - 6) : (t - 198);
        int dim = s >> 6;          // which spatial axis
        int f   = s & 63;          // frequency index
        const float* Lr = L + g * 9 + dim * 3;   // cart[e,k] = frac . L[g][k]
        const float* fr = frac + (size_t)e * 3;
        float cart = fr[0] * Lr[0] + fr[1] * Lr[1] + fr[2] * Lr[2];
        float xn = cart / lat[g * 6 + dim];
        float v  = xn * (6.283185307179586f * (float)f);
        out = (_Float16)((t < 198) ? sinf(v) : cosf(v));
    } else {
        out = (_Float16)0.0f;
    }
    tail[(size_t)e * TAILW + t] = out;
}

// Per-node in-degree counts
__global__ void cnt_kernel(const int* __restrict__ idx0, float* __restrict__ cnt) {
    int e = blockIdx.x * blockDim.x + threadIdx.x;
    if (e < NE) atomicAdd(&cnt[idx0[e]], 1.0f);
}

// agg / max(cnt,1) -> f16 into nodecat[:, 512:1024]
__global__ void aggfin_kernel(const float* __restrict__ agg, const float* __restrict__ cnt,
                              _Float16* __restrict__ nodecat) {
    long long i = (long long)blockIdx.x * blockDim.x + threadIdx.x;
    if (i >= (long long)NN * HID) return;
    int node = (int)(i >> 9);
    int c    = (int)(i & 511);
    nodecat[(size_t)node * 1024 + 512 + c] =
        (_Float16)(agg[i] / fmaxf(cnt[node], 1.0f));
}

// ---------------------------------------------------------------------------
// Tiled WMMA GEMM: C[M][512] = silu(A[M][K] * B[K][512] + bias), f16 in / f32 acc.
// Double-buffered LDS pipeline: fetch chunk k+1 into regs, WMMA on buf[k&1],
// stage regs into buf[1-(k&1)], single barrier per chunk.
// All per-thread source addressing is a 32-bit byte offset against the uniform
// kernel-arg base pointers (SGPR saddr + VGPR voffset loads, low VGPR cost).
// ---------------------------------------------------------------------------
struct GemmArgs {
    const float*    ef32;    // mode1
    const _Float16* h16;     // mode1: nodecat rows, stride 1024
    const _Float16* tail16;  // mode1
    const _Float16* a16;     // modes 2/3/4
    int             aStride;
    const int*      idx0;    // modes 1/2
    const int*      idx1;    // mode 1
    const _Float16* BT;      // [512][K]
    const float*    bias;    // [512]
    int             M;
    int             K;       // multiple of 32
    _Float16*       outH;    // modes 1/3
    float*          outF;    // mode 4
    const float*    addF;    // mode 4 residual
    float*          agg;     // mode 2
};

struct PF { uint4 a0, a1, b0, b1; };

__device__ __forceinline__ uint4 cvt8_f32_f16(const float* p) {
    float4 f0 = *(const float4*)p;
    float4 f1 = *(const float4*)(p + 4);
    H8 t;
    t.h[0] = (_Float16)f0.x; t.h[1] = (_Float16)f0.y;
    t.h[2] = (_Float16)f0.z; t.h[3] = (_Float16)f0.w;
    t.h[4] = (_Float16)f1.x; t.h[5] = (_Float16)f1.y;
    t.h[6] = (_Float16)f1.z; t.h[7] = (_Float16)f1.w;
    return t.q;
}

template <int MODE>
__device__ __forceinline__ PF fetch_chunk(const GemmArgs& g, int k0,
                                          int oEf0, int oEf1, int oH00, int oH01,
                                          int oH10, int oH11, int oT0, int oT1,
                                          int oA0, int oA1, int oB0, int oB1) {
    PF r;
    if (MODE == 1) {
        if (k0 < 512) {               // edge_feats: f32 -> f16 on the fly
            const char* base = (const char*)g.ef32 + (size_t)(unsigned)(k0 * 4);
            r.a0 = cvt8_f32_f16((const float*)(base + (unsigned)oEf0));
            r.a1 = cvt8_f32_f16((const float*)(base + (unsigned)oEf1));
        } else if (k0 < 1024) {
            const char* base = (const char*)g.h16 + (size_t)(unsigned)((k0 - 512) * 2);
            r.a0 = *(const uint4*)(base + (unsigned)oH00);
            r.a1 = *(const uint4*)(base + (unsigned)oH01);
        } else if (k0 < 1536) {
            const char* base = (const char*)g.h16 + (size_t)(unsigned)((k0 - 1024) * 2);
            r.a0 = *(const uint4*)(base + (unsigned)oH10);
            r.a1 = *(const uint4*)(base + (unsigned)oH11);
        } else {
            const char* base = (const char*)g.tail16 + (size_t)(unsigned)((k0 - 1536) * 2);
            r.a0 = *(const uint4*)(base + (unsigned)oT0);
            r.a1 = *(const uint4*)(base + (unsigned)oT1);
        }
    } else {
        const char* base = (const char*)g.a16 + (size_t)(unsigned)(k0 * 2);
        r.a0 = *(const uint4*)(base + (unsigned)oA0);
        r.a1 = *(const uint4*)(base + (unsigned)oA1);
    }
    const char* bbase = (const char*)g.BT + (size_t)(unsigned)(k0 * 2);
    r.b0 = *(const uint4*)(bbase + (unsigned)oB0);
    r.b1 = *(const uint4*)(bbase + (unsigned)oB1);
    return r;
}

template <int MODE>
__global__ __launch_bounds__(256, 1)
void gemm_kernel(GemmArgs g) {
    __shared__ _Float16 At[2][BM][LDST];
    __shared__ _Float16 Bt[2][BN][LDST];

    const int tid  = threadIdx.x;
    const int lane = tid & 31;
    const int wid  = tid >> 5;
    const int wm   = wid >> 2;      // 0..1
    const int wn   = wid & 3;       // 0..3
    const int tM   = blockIdx.y;
    const int tN   = blockIdx.x;
    const int rowBase = tM * BM;

    // ---- per-thread staging slots: rows {row0, row0+64}, fixed across K ----
    const int row0 = tid >> 2;          // 0..63
    const int row1 = row0 + 64;
    const int seg  = (tid & 3) * 8;     // 0,8,16,24 (halves)

    int e0 = rowBase + row0; if (e0 >= g.M) e0 = g.M - 1;
    int e1 = rowBase + row1; if (e1 >= g.M) e1 = g.M - 1;

    // 32-bit byte offsets against uniform bases (all buffers < 256 MB)
    int oEf0 = 0, oEf1 = 0, oH00 = 0, oH01 = 0, oH10 = 0, oH11 = 0;
    int oT0 = 0, oT1 = 0, oA0 = 0, oA1 = 0;
    if (MODE == 1) {
        oEf0 = (e0 * 512 + seg) * 4;
        oEf1 = (e1 * 512 + seg) * 4;
        oH00 = (g.idx0[e0] * 1024 + seg) * 2;
        oH01 = (g.idx0[e1] * 1024 + seg) * 2;
        oH10 = (g.idx1[e0] * 1024 + seg) * 2;
        oH11 = (g.idx1[e1] * 1024 + seg) * 2;
        oT0  = (e0 * TAILW + seg) * 2;
        oT1  = (e1 * TAILW + seg) * 2;
    } else {
        oA0 = (e0 * g.aStride + seg) * 2;
        oA1 = (e1 * g.aStride + seg) * 2;
    }
    const int oB0 = ((tN * BN + row0) * g.K + seg) * 2;
    const int oB1 = ((tN * BN + row1) * g.K + seg) * 2;

    FAcc acc[4][2];
    #pragma unroll
    for (int i = 0; i < 4; ++i)
        #pragma unroll
        for (int j = 0; j < 2; ++j)
            #pragma unroll
            for (int r = 0; r < 8; ++r) acc[i][j].f[r] = 0.0f;

    const int nChunks = g.K / BK;

    // prologue: chunk 0 -> buf 0
    {
        PF pf = fetch_chunk<MODE>(g, 0, oEf0, oEf1, oH00, oH01, oH10, oH11,
                                  oT0, oT1, oA0, oA1, oB0, oB1);
        *(uint4*)&At[0][row0][seg] = pf.a0;
        *(uint4*)&At[0][row1][seg] = pf.a1;
        *(uint4*)&Bt[0][row0][seg] = pf.b0;
        *(uint4*)&Bt[0][row1][seg] = pf.b1;
    }
    __syncthreads();

    for (int kc = 0; kc < nChunks; ++kc) {
        const int cur = kc & 1;
        const bool more = (kc + 1) < nChunks;

        PF pf;
        if (more)
            pf = fetch_chunk<MODE>(g, (kc + 1) * BK, oEf0, oEf1, oH00, oH01,
                                   oH10, oH11, oT0, oT1, oA0, oA1, oB0, oB1);

        // ---- WMMA fragments from LDS buf[cur] ----
        HFrag a[4], b[2];
        const int am  = lane & 15;
        const int akb = (lane >> 4) * 8;    // A: K in {kb..kb+7, kb+16..kb+23}
        #pragma unroll
        for (int i = 0; i < 4; ++i) {
            const _Float16* p = &At[cur][wm * 64 + i * 16 + am][akb];
            a[i].q[0] = *(const uint4*)p;
            a[i].q[1] = *(const uint4*)(p + 16);
        }
        const int bkb = (lane >> 4) * 8;    // B: K in {kb..kb+7, kb+16..kb+23}
        #pragma unroll
        for (int j = 0; j < 2; ++j) {
            const _Float16* p = &Bt[cur][wn * 32 + j * 16 + am][bkb];
            b[j].q[0] = *(const uint4*)p;
            b[j].q[1] = *(const uint4*)(p + 16);
        }
        #pragma unroll
        for (int i = 0; i < 4; ++i)
            #pragma unroll
            for (int j = 0; j < 2; ++j)
                acc[i][j].v = __builtin_amdgcn_wmma_f32_16x16x32_f16(
                    false, a[i].v, false, b[j].v, (short)0, acc[i][j].v, false, false);

        if (more) {                          // write-only to the other buffer
            const int nxt = 1 - cur;
            *(uint4*)&At[nxt][row0][seg] = pf.a0;
            *(uint4*)&At[nxt][row1][seg] = pf.a1;
            *(uint4*)&Bt[nxt][row0][seg] = pf.b0;
            *(uint4*)&Bt[nxt][row1][seg] = pf.b1;
        }
        __syncthreads();
    }

    // ---- epilogue: C layout lane->(M=r + 8*(lane>=16), N=lane&15) ----
    const int mb  = wm * 64 + (lane >> 4) * 8;
    const int nc0 = tN * BN + wn * 32 + (lane & 15);
    const float bias0 = g.bias[nc0];
    const float bias1 = g.bias[nc0 + 16];
    #pragma unroll
    for (int i = 0; i < 4; ++i) {
        #pragma unroll
        for (int j = 0; j < 2; ++j) {
            const int ncol = nc0 + j * 16;
            const float bj = (j == 0) ? bias0 : bias1;
            #pragma unroll
            for (int r = 0; r < 8; ++r) {
                const int m = rowBase + mb + i * 16 + r;
                if (m < g.M) {
                    float s = silu_f(acc[i][j].f[r] + bj);
                    if (MODE == 1) {
                        g.outH[(size_t)m * HID + ncol] = (_Float16)s;
                    } else if (MODE == 2) {
                        atomicAdd(&g.agg[(size_t)g.idx0[m] * HID + ncol], s);
                    } else if (MODE == 3) {
                        g.outH[(size_t)m * HID + ncol] = (_Float16)s;
                    } else {
                        g.outF[(size_t)m * HID + ncol] =
                            g.addF[(size_t)m * HID + ncol] + s;
                    }
                }
            }
        }
    }
}

// ---------------------------------------------------------------------------
// Host side
// ---------------------------------------------------------------------------
extern "C" void kernel_launch(void* const* d_in, const int* in_sizes, int n_in,
                              void* d_out, int out_size, void* d_ws, size_t ws_size,
                              hipStream_t stream) {
    (void)in_sizes; (void)n_in; (void)out_size; (void)ws_size;
    const float* node_features = (const float*)d_in[0];
    const float* lattices      = (const float*)d_in[1];
    const float* frac_diff     = (const float*)d_in[2];
    const float* edge_feats    = (const float*)d_in[3];
    const float* ln_gamma      = (const float*)d_in[4];
    const float* ln_beta       = (const float*)d_in[5];
    const float* eW1           = (const float*)d_in[6];
    const float* eb1           = (const float*)d_in[7];
    const float* eW2           = (const float*)d_in[8];
    const float* eb2           = (const float*)d_in[9];
    const float* nW1           = (const float*)d_in[10];
    const float* nb1           = (const float*)d_in[11];
    const float* nW2           = (const float*)d_in[12];
    const float* nb2           = (const float*)d_in[13];
    const int*   edge_index    = (const int*)d_in[14];
    const int*   edge2graph    = (const int*)d_in[15];
    float* out = (float*)d_out;

    const int* idx0 = edge_index;            // edge_index[0]
    const int* idx1 = edge_index + NE;       // edge_index[1]

    // workspace layout
    char* ws = (char*)d_ws;
    size_t off = 0;
    auto take = [&](size_t bytes) {
        char* p = ws + off;
        off = (off + bytes + 255) & ~(size_t)255;
        return p;
    };
    _Float16* W1T     = (_Float16*)take((size_t)512 * KPAD1 * 2);
    _Float16* W2T     = (_Float16*)take((size_t)512 * 512 * 2);
    _Float16* W3T     = (_Float16*)take((size_t)512 * 1024 * 2);
    _Float16* W4T     = (_Float16*)take((size_t)512 * 512 * 2);
    float*    Lmat    = (float*)take((size_t)NG * 9 * 4);
    _Float16* nodecat = (_Float16*)take((size_t)NN * 1024 * 2);  // [h16 | agg16]
    _Float16* tail16  = (_Float16*)take((size_t)NE * TAILW * 2);
    _Float16* m1      = (_Float16*)take((size_t)NE * HID * 2);
    _Float16* nh1     = (_Float16*)take((size_t)NN * HID * 2);
    float*    agg     = (float*)take((size_t)NN * HID * 4);
    float*    cnt     = (float*)take((size_t)NN * 4);

    // ---- prep ----
    zerof_kernel<<<(NN * HID + 255) / 256, 256, 0, stream>>>(agg, (long long)NN * HID);
    zerof_kernel<<<(NN + 255) / 256, 256, 0, stream>>>(cnt, NN);

    packw_kernel<<<(512 * KPAD1 + 255) / 256, 256, 0, stream>>>(eW1, W1T, 1926, KPAD1);
    packw_kernel<<<(512 * 512 + 255) / 256, 256, 0, stream>>>(eW2, W2T, 512, 512);
    packw_kernel<<<(512 * 1024 + 255) / 256, 256, 0, stream>>>(nW1, W3T, 1024, 1024);
    packw_kernel<<<(512 * 512 + 255) / 256, 256, 0, stream>>>(nW2, W4T, 512, 512);

    lattice_kernel<<<1, 64, 0, stream>>>(lattices, Lmat);
    layernorm_kernel<<<(NN + 7) / 8, 256, 0, stream>>>(node_features, ln_gamma, ln_beta,
                                                       nodecat, NN);
    tail_kernel<<<(int)(((long long)NE * TAILW + 255) / 256), 256, 0, stream>>>(
        frac_diff, lattices, Lmat, edge2graph, tail16);
    cnt_kernel<<<(NE + 255) / 256, 256, 0, stream>>>(idx0, cnt);

    // ---- GEMM 1: edge MLP layer 1 (fused gather+concat) ----
    {
        GemmArgs a{};
        a.ef32 = edge_feats; a.h16 = nodecat; a.tail16 = tail16;
        a.idx0 = idx0; a.idx1 = idx1;
        a.BT = W1T; a.bias = eb1; a.M = NE; a.K = KPAD1; a.outH = m1;
        gemm_kernel<1><<<dim3(4, (NE + BM - 1) / BM), 256, 0, stream>>>(a);
    }
    // ---- GEMM 2: edge MLP layer 2, epilogue scatters into agg ----
    {
        GemmArgs a{};
        a.a16 = m1; a.aStride = HID; a.idx0 = idx0;
        a.BT = W2T; a.bias = eb2; a.M = NE; a.K = 512; a.agg = agg;
        gemm_kernel<2><<<dim3(4, (NE + BM - 1) / BM), 256, 0, stream>>>(a);
    }
    // ---- mean aggregate -> nodecat[:, 512:1024] ----
    aggfin_kernel<<<(NN * HID + 255) / 256, 256, 0, stream>>>(agg, cnt, nodecat);

    // ---- GEMM 3: node MLP layer 1 over [h | agg] ----
    {
        GemmArgs a{};
        a.a16 = nodecat; a.aStride = 1024;
        a.BT = W3T; a.bias = nb1; a.M = NN; a.K = 1024; a.outH = nh1;
        gemm_kernel<3><<<dim3(4, (NN + BM - 1) / BM), 256, 0, stream>>>(a);
    }
    // ---- GEMM 4: node MLP layer 2 + residual -> d_out ----
    {
        GemmArgs a{};
        a.a16 = nh1; a.aStride = HID;
        a.BT = W4T; a.bias = nb2; a.M = NN; a.K = 512;
        a.outF = out; a.addF = node_features;
        gemm_kernel<4><<<dim3(4, (NN + BM - 1) / BM), 256, 0, stream>>>(a);
    }
}
